// AgentAttention_77532749627633
// MI455X (gfx1250) — matrix operand order repaired and checked by reading, not compiled, not run on hardware
//
#include <hip/hip_runtime.h>
#include <hip/hip_bf16.h>

// ---------------------------------------------------------------------------
// Types for CDNA5 WMMA
// ---------------------------------------------------------------------------
typedef __attribute__((ext_vector_type(16))) __bf16 v16bf;
typedef __attribute__((ext_vector_type(8)))  float  v8f;
typedef __attribute__((ext_vector_type(4)))  int    i32x4;

union Frag {
    v16bf          v;
    uint4          u4[2];
    unsigned short s[16];
};

__device__ __forceinline__ v8f wmma_bf16(v16bf a, v16bf b, v8f c) {
    // (neg_a, A, neg_b, B, c_mod, C, reuse_a, reuse_b)
    return __builtin_amdgcn_wmma_f32_16x16x32_bf16(false, a, false, b, (short)0, c, false, false);
}

__device__ __forceinline__ unsigned short f2bf(float f) {
    unsigned int u = __float_as_uint(f);
    u += 0x7fffu + ((u >> 16) & 1u);   // round-to-nearest-even
    return (unsigned short)(u >> 16);
}
__device__ __forceinline__ float bf2f(unsigned short h) {
    return __uint_as_float(((unsigned int)h) << 16);
}

// ---------------------------------------------------------------------------
// CDNA5 async global->LDS copy (16 bytes), tracked by ASYNCcnt
// ---------------------------------------------------------------------------
__device__ __forceinline__ void async_cp16(const unsigned short* g, unsigned short* l) {
#if __has_builtin(__builtin_amdgcn_global_load_async_to_lds_b128)
    __builtin_amdgcn_global_load_async_to_lds_b128(
        (__attribute__((address_space(1))) i32x4*)(g),
        (__attribute__((address_space(3))) i32x4*)(l),
        0, 0);
#else
    unsigned int lv = (unsigned int)(size_t)(__attribute__((address_space(3))) void*)l;
    asm volatile("global_load_async_to_lds_b128 %0, %1, off"
                 :: "v"(lv), "v"(g) : "memory");
#endif
}

__device__ __forceinline__ void wait_async0() {
#if __has_builtin(__builtin_amdgcn_s_wait_asynccnt)
    __builtin_amdgcn_s_wait_asynccnt(0);
#else
    asm volatile("s_wait_asynccnt 0x0" ::: "memory");
#endif
}

// Problem constants
#define B_  64
#define N_  1024
#define C_  512
#define M_  16
#define BN_ (B_ * N_)          // 65536
#define SCALE_ 0.04419417382415922f   // 512^-0.5

// ---------------------------------------------------------------------------
// fp32 -> bf16 conversion (4 elements / thread)
// ---------------------------------------------------------------------------
__global__ __launch_bounds__(256) void cvt_f32_bf16(const float* __restrict__ src,
                                                    unsigned short* __restrict__ dst,
                                                    int n) {
    int i = (blockIdx.x * 256 + threadIdx.x) * 4;
    if (i + 3 < n) {
        float4 f = *(const float4*)(src + i);
        unsigned int p0 = (unsigned int)f2bf(f.x) | ((unsigned int)f2bf(f.y) << 16);
        unsigned int p1 = (unsigned int)f2bf(f.z) | ((unsigned int)f2bf(f.w) << 16);
        *(uint2*)(dst + i) = make_uint2(p0, p1);
    }
}

// ---------------------------------------------------------------------------
// GEMM: out[i,j] = sum_k A[i,k]*W[j,k] + bias[j]   (nn.Linear)
// A: [65536 x 512] bf16 row-major, W: [512 x 512] bf16 row-major.
// Block = 256 threads (8 waves), tile 128x128, K-step 32.
// Double-buffered LDS filled with async global->LDS copies (ASYNCcnt),
// overlapped with WMMA compute on the other buffer.
// ---------------------------------------------------------------------------
template <bool OUT_F32>
__global__ __launch_bounds__(256) void gemm_bf16(const unsigned short* __restrict__ A,
                                                 const unsigned short* __restrict__ W,
                                                 const float* __restrict__ bias,
                                                 void* __restrict__ Out) {
    __shared__ unsigned short As[2][128][40];   // padded: 80B rows, conflict-free
    __shared__ unsigned short Bs[2][128][40];

    const int tid  = threadIdx.x;
    const int lane = tid & 31;
    const int wave = tid >> 5;
    const int wm   = wave >> 1;              // 0..3 (32-row strip)
    const int wn   = wave & 1;               // 0..1 (64-col strip)
    const int cb   = (blockIdx.x & 3) * 128; // col base
    const int rb   = (blockIdx.x >> 2) * 128;// row base

    const int l15  = lane & 15;
    const int aoff = (lane >> 4) << 3;       // A frag K sub-offset: 0/8
    const int boff = (lane >> 4) << 4;       // unused; B frags also use aoff (0/8)

    v8f acc[2][4];
    for (int r = 0; r < 2; ++r)
        for (int c = 0; c < 4; ++c)
            acc[r][c] = (v8f){0.f,0.f,0.f,0.f,0.f,0.f,0.f,0.f};

    // async fill of one 128x32 double-tile (A and W), 4 b128s per thread
    auto prefetch = [&](int buf, int kc) {
        for (int i = tid; i < 512; i += 256) {
            int row = i >> 2, ch = (i & 3) * 8;
            async_cp16(A + (size_t)(rb + row) * C_ + kc + ch, &As[buf][row][ch]);
            async_cp16(W + (size_t)(cb + row) * C_ + kc + ch, &Bs[buf][row][ch]);
        }
    };

    prefetch(0, 0);
    wait_async0();
    __syncthreads();

    int buf = 0;
    for (int kc = 0; kc < C_; kc += 32) {
        if (kc + 32 < C_) prefetch(buf ^ 1, kc + 32);   // overlap with compute

        Frag a[2], b[4];
        for (int r = 0; r < 2; ++r) {
            int ar = wm * 32 + r * 16 + l15;
            a[r].u4[0] = *(const uint4*)&As[buf][ar][aoff];
            a[r].u4[1] = *(const uint4*)&As[buf][ar][16 + aoff];
        }
        for (int c = 0; c < 4; ++c) {
            int br = wn * 64 + c * 16 + l15;
            b[c].u4[0] = *(const uint4*)&Bs[buf][br][aoff];
            b[c].u4[1] = *(const uint4*)&Bs[buf][br][16 + aoff];
        }
        for (int r = 0; r < 2; ++r)
            for (int c = 0; c < 4; ++c)
                acc[r][c] = wmma_bf16(a[r].v, b[c].v, acc[r][c]);

        wait_async0();        // own async fills of buf^1 have landed
        __syncthreads();      // everyone done reading buf / filling buf^1
        buf ^= 1;
    }

    const int rshift = (lane >> 4) << 3;     // D row offset: 0/8
    for (int r = 0; r < 2; ++r)
        for (int c = 0; c < 4; ++c) {
            int col = cb + wn * 64 + c * 16 + l15;
            float bv = bias[col];
            for (int vr = 0; vr < 8; ++vr) {
                int row = rb + wm * 32 + r * 16 + vr + rshift;
                float v = acc[r][c][vr] + bv;
                if (OUT_F32) ((float*)Out)[(size_t)row * C_ + col] = v;
                else ((unsigned short*)Out)[(size_t)row * C_ + col] = f2bf(v);
            }
        }
}

// ---------------------------------------------------------------------------
// Agent pooling: Ab[b,m,c] = mean_{t<64} Qb[b, m*64+t, c]
// ---------------------------------------------------------------------------
__global__ __launch_bounds__(256) void pool_q(const unsigned short* __restrict__ Qb,
                                              unsigned short* __restrict__ Ab) {
    int i = blockIdx.x * 256 + threadIdx.x;      // over B*M*C = 524288
    int c = i & 511;
    int m = (i >> 9) & 15;
    int b = i >> 13;
    const unsigned short* p = Qb + ((size_t)(b * N_ + m * 64) * C_ + c);
    float s = 0.f;
    for (int t = 0; t < 64; ++t) s += bf2f(p[(size_t)t * C_]);
    Ab[i] = f2bf(s * (1.0f / 64.0f));
}

// ---------------------------------------------------------------------------
// Stage 1: agent_tokens[b,n,m] = softmax_m( scale * Q[b,n,:] . A[b,m,:] )
// One wave per (b, 16-token block); 16x16x512 via 16 WMMAs, softmax over M=16.
// ---------------------------------------------------------------------------
__global__ __launch_bounds__(32) void stage1(const unsigned short* __restrict__ Qb,
                                             const unsigned short* __restrict__ Ab,
                                             unsigned short* __restrict__ at) {
    const int b  = blockIdx.x >> 6;
    const int nb = blockIdx.x & 63;
    const int lane = threadIdx.x;
    const int l15  = lane & 15;
    const int aoff = (lane >> 4) << 3;
    const int boff = (lane >> 4) << 4;
    const unsigned short* qrow = Qb + ((size_t)(b * N_ + nb * 16 + l15)) * C_;
    const unsigned short* arow = Ab + ((size_t)(b * M_ + l15)) * C_;

    v8f acc = (v8f){0.f,0.f,0.f,0.f,0.f,0.f,0.f,0.f};
    for (int kc = 0; kc < C_; kc += 32) {
        Frag a, bb;
        a.u4[0]  = *(const uint4*)(qrow + kc + aoff);
        a.u4[1]  = *(const uint4*)(qrow + kc + 16 + aoff);
        bb.u4[0] = *(const uint4*)(arow + kc + aoff);
        bb.u4[1] = *(const uint4*)(arow + kc + 16 + aoff);
        acc = wmma_bf16(a.v, bb.v, acc);
    }

    __shared__ float L[16][17];
    const int rshift = (lane >> 4) << 3;
    for (int r = 0; r < 8; ++r) L[r + rshift][l15] = acc[r] * SCALE_;
    __syncthreads();

    if (lane < 16) {
        float mx = -1e30f;
        for (int j = 0; j < 16; ++j) mx = fmaxf(mx, L[lane][j]);
        float e[16], sum = 0.f;
        for (int j = 0; j < 16; ++j) { e[j] = __expf(L[lane][j] - mx); sum += e[j]; }
        float inv = 1.0f / sum;
        unsigned short* o = at + ((size_t)(b * N_ + nb * 16 + lane)) * 16;
        for (int j = 0; j < 16; ++j) o[j] = f2bf(e[j] * inv);
    }
}

// ---------------------------------------------------------------------------
// Stage 2a: logits2[b,m,n] = scale * A[b,m,:] . K[b,n,:]   (fp32)
// ---------------------------------------------------------------------------
__global__ __launch_bounds__(32) void stage2_logits(const unsigned short* __restrict__ Ab,
                                                    const unsigned short* __restrict__ Kb,
                                                    float* __restrict__ logits) {
    const int b  = blockIdx.x >> 6;
    const int nb = blockIdx.x & 63;
    const int lane = threadIdx.x;
    const int l15  = lane & 15;
    const int aoff = (lane >> 4) << 3;
    const int boff = (lane >> 4) << 4;
    const unsigned short* arow = Ab + ((size_t)(b * M_ + l15)) * C_;
    const unsigned short* krow = Kb + ((size_t)(b * N_ + nb * 16 + l15)) * C_;

    v8f acc = (v8f){0.f,0.f,0.f,0.f,0.f,0.f,0.f,0.f};
    for (int kc = 0; kc < C_; kc += 32) {
        Frag a, bb;
        a.u4[0]  = *(const uint4*)(arow + kc + aoff);
        a.u4[1]  = *(const uint4*)(arow + kc + 16 + aoff);
        bb.u4[0] = *(const uint4*)(krow + kc + aoff);
        bb.u4[1] = *(const uint4*)(krow + kc + 16 + aoff);
        acc = wmma_bf16(a.v, bb.v, acc);
    }
    const int rshift = (lane >> 4) << 3;
    for (int r = 0; r < 8; ++r) {
        int m = r + rshift;
        logits[((size_t)(b * M_ + m)) * N_ + nb * 16 + l15] = acc[r] * SCALE_;
    }
}

// ---------------------------------------------------------------------------
// Stage 2b: row softmax over n=1024, output bf16 probabilities
// ---------------------------------------------------------------------------
__global__ __launch_bounds__(256) void softmax_row(const float* __restrict__ logits,
                                                   unsigned short* __restrict__ probs) {
    const float* row = logits + (size_t)blockIdx.x * N_;
    unsigned short* out = probs + (size_t)blockIdx.x * N_;
    const int t = threadIdx.x;
    __shared__ float red[256];

    float mx = -1e30f;
    for (int i = t; i < N_; i += 256) mx = fmaxf(mx, row[i]);
    red[t] = mx; __syncthreads();
    for (int s = 128; s > 0; s >>= 1) { if (t < s) red[t] = fmaxf(red[t], red[t + s]); __syncthreads(); }
    mx = red[0]; __syncthreads();

    float sum = 0.f;
    for (int i = t; i < N_; i += 256) sum += __expf(row[i] - mx);
    red[t] = sum; __syncthreads();
    for (int s = 128; s > 0; s >>= 1) { if (t < s) red[t] += red[t + s]; __syncthreads(); }
    float inv = 1.0f / red[0];

    for (int i = t; i < N_; i += 256) out[i] = f2bf(__expf(row[i] - mx) * inv);
}

// ---------------------------------------------------------------------------
// Stage 2c: agent_features[b,m,c] = sum_n probs[b,m,n] * V[b,n,c]
// One wave per (b, 16-col tile); K = 1024.
// ---------------------------------------------------------------------------
__global__ __launch_bounds__(32) void stage2_av(const unsigned short* __restrict__ probs,
                                                const unsigned short* __restrict__ Vb,
                                                unsigned short* __restrict__ af) {
    const int b  = blockIdx.x >> 5;
    const int ct = blockIdx.x & 31;
    const int lane = threadIdx.x;
    const int l15  = lane & 15;
    const int aoff = (lane >> 4) << 3;
    const int c = ct * 16 + l15;
    const unsigned short* prow = probs + ((size_t)(b * M_ + l15)) * N_;

    v8f acc = (v8f){0.f,0.f,0.f,0.f,0.f,0.f,0.f,0.f};
    for (int kc = 0; kc < N_; kc += 32) {
        Frag a, bb;
        a.u4[0] = *(const uint4*)(prow + kc + 2 * aoff);
        a.u4[1] = *(const uint4*)(prow + kc + 2 * aoff + 8);
        int ks = kc + ((lane >> 4) << 4);
        for (int i = 0; i < 16; ++i)
            bb.s[i] = Vb[((size_t)(b * N_ + ks + i)) * C_ + c];   // V^T gather
        acc = wmma_bf16(a.v, bb.v, acc);
    }
    const int rshift = (lane >> 4) << 3;
    for (int r = 0; r < 8; ++r)
        af[((size_t)(b * M_ + r + rshift)) * C_ + c] = f2bf(acc[r]);
}

// ---------------------------------------------------------------------------
// Combine: agent_attn[b,n,c] = sum_{m<16} at[b,n,m] * af[b,m,c]
// K=16 zero-padded into one 16x16x32 WMMA per tile. 8 waves/block, 4 tiles/wave.
// ---------------------------------------------------------------------------
__global__ __launch_bounds__(256) void combine(const unsigned short* __restrict__ at,
                                               const unsigned short* __restrict__ af,
                                               unsigned short* __restrict__ attn) {
    const int b  = blockIdx.x >> 6;
    const int nb = blockIdx.x & 63;
    const int tid  = threadIdx.x;
    const int lane = tid & 31;
    const int w    = tid >> 5;
    const int l15  = lane & 15;
    const int aoff = (lane >> 4) << 3;

    Frag a;
    const unsigned short* trow = at + ((size_t)(b * N_ + nb * 16 + l15)) * 16;
    for (int i = 0; i < 8; ++i)  a.s[i] = trow[aoff + i];
    for (int i = 8; i < 16; ++i) a.s[i] = 0;                 // K=16..31 pad

    const int rshift = aoff;
    for (int cc = 0; cc < 4; ++cc) {
        int c = (w + cc * 8) * 16 + l15;
        Frag bb;
        for (int i = 0; i < 8; ++i)
            bb.s[i] = af[((size_t)(b * M_ + aoff + i)) * C_ + c];
        for (int i = 8; i < 16; ++i) bb.s[i] = 0;            // K=16..31 pad
        v8f acc = (v8f){0.f,0.f,0.f,0.f,0.f,0.f,0.f,0.f};
        acc = wmma_bf16(a.v, bb.v, acc);
        for (int r = 0; r < 8; ++r)
            attn[((size_t)(b * N_ + nb * 16 + r + rshift)) * C_ + c] = f2bf(acc[r]);
    }
}

// ---------------------------------------------------------------------------
// Host orchestration
// ---------------------------------------------------------------------------
extern "C" void kernel_launch(void* const* d_in, const int* in_sizes, int n_in,
                              void* d_out, int out_size, void* d_ws, size_t ws_size,
                              hipStream_t stream) {
    const float* x  = (const float*)d_in[0];
    const float* Wq = (const float*)d_in[1];
    const float* bq = (const float*)d_in[2];
    const float* Wk = (const float*)d_in[3];
    const float* bk = (const float*)d_in[4];
    const float* Wv = (const float*)d_in[5];
    const float* bv = (const float*)d_in[6];
    const float* Wo = (const float*)d_in[7];
    const float* bo = (const float*)d_in[8];
    float* out = (float*)d_out;

    // workspace carve-up (bytes)
    char* w = (char*)d_ws;
    unsigned short* xb   = (unsigned short*)(w);                         // 64 MB (reused as attn)
    unsigned short* Wqb  = (unsigned short*)(w + 67108864);
    unsigned short* Wkb  = (unsigned short*)(w + 67633152);
    unsigned short* Wvb  = (unsigned short*)(w + 68157440);
    unsigned short* Wob  = (unsigned short*)(w + 68681728);
    unsigned short* Qb   = (unsigned short*)(w + 69206016);              // 64 MB
    unsigned short* Kb   = (unsigned short*)(w + 136314880);             // 64 MB
    unsigned short* Vb   = (unsigned short*)(w + 203423744);             // 64 MB
    unsigned short* Ab   = (unsigned short*)(w + 270532608);             // 1 MB
    unsigned short* at   = (unsigned short*)(w + 271581184);             // 2 MB
    float*          lg2  = (float*)         (w + 273678336);             // 4 MB
    unsigned short* prb  = (unsigned short*)(w + 277872640);             // 2 MB
    unsigned short* af   = (unsigned short*)(w + 279969792);             // 1 MB
    unsigned short* attn = xb;                                           // alias (xb dead by then)

    // 1) fp32 -> bf16
    cvt_f32_bf16<<<(BN_ * C_) / 1024, 256, 0, stream>>>(x,  xb,  BN_ * C_);
    cvt_f32_bf16<<<(C_ * C_) / 1024, 256, 0, stream>>>(Wq, Wqb, C_ * C_);
    cvt_f32_bf16<<<(C_ * C_) / 1024, 256, 0, stream>>>(Wk, Wkb, C_ * C_);
    cvt_f32_bf16<<<(C_ * C_) / 1024, 256, 0, stream>>>(Wv, Wvb, C_ * C_);
    cvt_f32_bf16<<<(C_ * C_) / 1024, 256, 0, stream>>>(Wo, Wob, C_ * C_);

    // 2) Q/K/V projections (bf16 out)
    const int gemm_blocks = (BN_ / 128) * (C_ / 128);   // 2048
    gemm_bf16<false><<<gemm_blocks, 256, 0, stream>>>(xb, Wqb, bq, Qb);
    gemm_bf16<false><<<gemm_blocks, 256, 0, stream>>>(xb, Wkb, bk, Kb);
    gemm_bf16<false><<<gemm_blocks, 256, 0, stream>>>(xb, Wvb, bv, Vb);

    // 3) agent pooling
    pool_q<<<(B_ * M_ * C_) / 256, 256, 0, stream>>>(Qb, Ab);

    // 4) agent_tokens = softmax_m(Q A^T)
    stage1<<<B_ * (N_ / 16), 32, 0, stream>>>(Qb, Ab, at);

    // 5) agent_features = softmax_n(A K^T) V
    stage2_logits<<<B_ * (N_ / 16), 32, 0, stream>>>(Ab, Kb, lg2);
    softmax_row<<<B_ * M_, 256, 0, stream>>>(lg2, prb);
    stage2_av<<<B_ * (C_ / 16), 32, 0, stream>>>(prb, Vb, af);

    // 6) combine (overwrites xb region with agent_attn bf16)
    combine<<<B_ * (N_ / 16), 256, 0, stream>>>(at, af, attn);

    // 7) output projection -> fp32 d_out
    gemm_bf16<true><<<gemm_blocks, 256, 0, stream>>>(attn, Wob, bo, (void*)out);
}
